// MultiheadAttention_10247791968450
// MI455X (gfx1250) — matrix operand order repaired and checked by reading, not compiled, not run on hardware
//
#include <hip/hip_runtime.h>

// ---------------------------------------------------------------------------
// MHA for MI455X (gfx1250, wave32).  v_wmma_f32_16x16x32_f16 everywhere,
// f16 operands staged in LDS via the CDNA5 async path
// (global_load_async_to_lds_b128 + s_wait_asynccnt).  GEMMs use 32x64
// register tiles per wave (8 WMMAs per staging round).  Flash-attention
// inner loop with online softmax in the S^T = K·Q^T layout (score frags
// feed the P·V WMMA with zero cross-lane movement).
// ---------------------------------------------------------------------------

typedef __attribute__((ext_vector_type(16))) _Float16 v16h;
typedef __attribute__((ext_vector_type(8)))  _Float16 v8h;
typedef __attribute__((ext_vector_type(4)))  _Float16 v4h;
typedef __attribute__((ext_vector_type(8)))  float    v8f;

static constexpr int cB  = 4;
static constexpr int cS  = 2048;
static constexpr int cD  = 1024;
static constexpr int cH  = 16;
static constexpr int cDK = 64;

__device__ __forceinline__ v8f wmma_f16(v16h a, v16h b, v8f c) {
  return __builtin_amdgcn_wmma_f32_16x16x32_f16(false, a, false, b, (short)0, c,
                                                false, false);
}

__device__ __forceinline__ v16h cat8(v8h a, v8h b) {
  return __builtin_shufflevector(a, b, 0, 1, 2, 3, 4, 5, 6, 7, 8, 9, 10, 11,
                                 12, 13, 14, 15);
}

// 8 contiguous f32 -> 8 f16
__device__ __forceinline__ v8h ldcvt8(const float* p) {
  const float4* p4 = reinterpret_cast<const float4*>(p);
  float4 a = p4[0], b = p4[1];
  v8h r;
  r[0] = (_Float16)a.x; r[1] = (_Float16)a.y;
  r[2] = (_Float16)a.z; r[3] = (_Float16)a.w;
  r[4] = (_Float16)b.x; r[5] = (_Float16)b.y;
  r[6] = (_Float16)b.z; r[7] = (_Float16)b.w;
  return r;
}

// 8 contiguous f16 (16B aligned) -> one b128 load (global or LDS)
__device__ __forceinline__ v8h ldh8(const _Float16* p) {
  return *reinterpret_cast<const v8h*>(p);
}

// CDNA5 async DMA: 16 bytes/lane, memory -> LDS, tracked by ASYNCcnt.
__device__ __forceinline__ void async_copy16(unsigned int lds_off, const void* g) {
  asm volatile("global_load_async_to_lds_b128 %0, %1, off"
               :: "v"(lds_off), "v"((unsigned long long)(uintptr_t)g)
               : "memory");
}
__device__ __forceinline__ void wait_async() {
  asm volatile("s_wait_asynccnt 0x0" ::: "memory");
}
__device__ __forceinline__ unsigned int lds_off_of(const void* p) {
  return (unsigned int)(uintptr_t)p;   // LDS aperture: offset lives in addr[31:0]
}

// ---------------------------------------------------------------------------
// Elementwise f32 -> f16 (weights pre-pass)
// ---------------------------------------------------------------------------
__global__ __launch_bounds__(256) void cvt_kernel(const float* __restrict__ x,
                                                  _Float16* __restrict__ o,
                                                  int n8) {
  int i = blockIdx.x * 256 + threadIdx.x;
  if (i < n8) {
    v8h h = ldcvt8(x + (size_t)i * 8);
    *reinterpret_cast<v8h*>(o + (size_t)i * 8) = h;
  }
}

// ---------------------------------------------------------------------------
// Head projection: out[m,n] = sum_k X[m,k]*W[n,k] + bias[n].
// Block = 8 waves sharing a 32-row M tile; each wave owns 64 N columns
// (2x4 = 8 f32 accumulators -> 8 WMMAs per k-step).  A (f32) staged through
// LDS with one cvt per element per block; B (pre-cvt f16) async-DMA'd into
// per-wave LDS slices.  mode 0: f16 [B,H,S,DK]; mode 1: f16 [B,H,DK,S].
// ---------------------------------------------------------------------------
__global__ __launch_bounds__(256) void proj16_kernel(const float* __restrict__ X,
                                                     const _Float16* __restrict__ Wh,
                                                     const float* __restrict__ bias,
                                                     _Float16* __restrict__ out,
                                                     int mode) {
  __shared__ _Float16 sA[32 * 32];            // [row][k]           2 KB
  __shared__ _Float16 sB[8 * 4 * 16 * 32];    // wave x ntile x [n][k], 32 KB

  const int tid  = threadIdx.x;
  const int wv   = tid >> 5;
  const int lane = tid & 31;
  const int hl   = lane >> 4;
  const int ln   = lane & 15;
  const int mt   = blockIdx.x >> 1;           // 0..255
  const int ngrp = blockIdx.x & 1;            // 0..1
  const int m0   = mt * 32;
  const int n0   = ngrp * 512 + wv * 64;      // wave's 64-column slice

  const unsigned int boff = lds_off_of(sB) + wv * 4096;
  _Float16* sBw = sB + wv * 2048;

  const int arow = tid >> 3;                  // 0..31
  const int acol = (tid & 7) * 4;             // 0..28
  const float* xrow = X + (size_t)(m0 + arow) * cD + acol;
  const char*  wb   = (const char*)Wh;

  v8f c[2][4];
#pragma unroll
  for (int mi = 0; mi < 2; ++mi)
#pragma unroll
    for (int j = 0; j < 4; ++j)
#pragma unroll
      for (int i = 0; i < 8; ++i) c[mi][j][i] = 0.f;

  for (int k0 = 0; k0 < cD; k0 += 32) {
    __syncthreads();                          // previous reads complete
    // stage A: 32x32 f32 -> f16 (4 elements per thread)
    {
      float4 xv = *reinterpret_cast<const float4*>(xrow + k0);
      v4h h;
      h[0] = (_Float16)xv.x; h[1] = (_Float16)xv.y;
      h[2] = (_Float16)xv.z; h[3] = (_Float16)xv.w;
      *reinterpret_cast<v4h*>(sA + arow * 32 + acol) = h;
    }
    // stage B: 4 n-tiles x 1KB, async DMA into this wave's private slice
#pragma unroll
    for (int j = 0; j < 4; ++j)
#pragma unroll
      for (int i = 0; i < 2; ++i) {
        int cch = i * 32 + lane, row = cch >> 2, off = (cch & 3) * 16;
        async_copy16(boff + j * 1024 + row * 64 + off,
                     wb + ((size_t)(n0 + j * 16 + row) * cD + k0) * 2 + off);
      }
    wait_async();
    __syncthreads();

    v16h Af[2], Bf[4];
#pragma unroll
    for (int mi = 0; mi < 2; ++mi) {
      const _Float16* ar = sA + (mi * 16 + ln) * 32;
      Af[mi] = cat8(ldh8(ar + hl * 8), ldh8(ar + 16 + hl * 8));
    }
#pragma unroll
    for (int j = 0; j < 4; ++j) {
      const _Float16* br = sBw + j * 512 + ln * 32;
      Bf[j] = cat8(ldh8(br + hl * 8), ldh8(br + 16 + hl * 8));
    }
#pragma unroll
    for (int mi = 0; mi < 2; ++mi)
#pragma unroll
      for (int j = 0; j < 4; ++j)
        c[mi][j] = wmma_f16(Af[mi], Bf[j], c[mi][j]);
  }

#pragma unroll
  for (int j = 0; j < 4; ++j) {
    const int   n  = n0 + j * 16 + ln;
    const float bv = bias[n];
    const int   h  = n >> 6;
    const int   e  = n & 63;
#pragma unroll
    for (int mi = 0; mi < 2; ++mi) {
#pragma unroll
      for (int r = 0; r < 8; ++r) {
        const int m = m0 + mi * 16 + r + 8 * hl;
        const int b = m >> 11;
        const int s = m & 2047;
        const float val = c[mi][j][r] + bv;
        size_t idx;
        if (mode == 0)
          idx = ((size_t)(b * cH + h) * cS + s) * cDK + e;   // [B,H,S,DK]
        else
          idx = ((size_t)(b * cH + h) * cDK + e) * cS + s;   // [B,H,DK,S]
        out[idx] = (_Float16)val;
      }
    }
  }
}

// ---------------------------------------------------------------------------
// Flash attention.  Block = 8 waves sharing one (b,h); per 32-key step the
// K block (4KB) and transposed-V block (4KB) are async-DMA'd into LDS once
// and consumed by all 8 waves.
// ---------------------------------------------------------------------------
__global__ __launch_bounds__(256) void flash_attn_kernel(
    const _Float16* __restrict__ qh,   // [B*H, S, DK]
    const _Float16* __restrict__ kh,   // [B*H, S, DK]
    const _Float16* __restrict__ vt,   // [B*H, DK, S]
    const unsigned char* __restrict__ kpm,  // [B, S] bool
    const unsigned char* __restrict__ am,   // [S, S] bool
    _Float16* __restrict__ y)          // [B, S, D]
{
  __shared__ _Float16 sK[32 * 64];     // [key][dk]   4 KB
  __shared__ _Float16 sV[64 * 32];     // [dk][key]   4 KB

  const int wv   = threadIdx.x >> 5;
  const int lane = threadIdx.x & 31;
  const int hl   = lane >> 4;
  const int ln   = lane & 15;
  const int w    = blockIdx.x * 8 + wv;
  const int qt   = w & 127;
  const int bh   = w >> 7;             // uniform across the block
  const int b    = bh >> 4, h = bh & 15;
  const int qb   = qt * 16;
  const int qg   = qb + ln;

  const unsigned int kloff = lds_off_of(sK);
  const unsigned int vloff = lds_off_of(sV);

  // Q as B-fragments (32 d-rows x 16 q-cols), loaded once from global.
  v16h Bq[2];
#pragma unroll
  for (int i = 0; i < 2; ++i) {
    const _Float16* p = qh + ((size_t)bh * cS + qg) * cDK + i * 32 + hl * 8;
    Bq[i] = cat8(ldh8(p), ldh8(p + 16));
  }

  v8f o[4];
#pragma unroll
  for (int f = 0; f < 4; ++f)
#pragma unroll
    for (int r = 0; r < 8; ++r) o[f][r] = 0.f;

  float mi = -1e30f, li = 0.f;
  const char* kbase = (const char*)(kh + (size_t)bh * cS * cDK);
  const char* vbase = (const char*)(vt + (size_t)bh * cDK * cS);
  const unsigned char* kpr = kpm + (size_t)b * cS;
  const unsigned char* amr = am + (size_t)qg * cS;

  for (int kb = 0; kb < cS; kb += 32) {
    // ---- async stage K (contiguous 4KB) and V (64 rows x 64B) ----
    if (wv == 0) {
#pragma unroll
      for (int i = 0; i < 8; ++i) {
        int cch = i * 32 + lane;
        async_copy16(kloff + cch * 16, kbase + (size_t)kb * cDK * 2 + cch * 16);
      }
    } else if (wv == 1) {
#pragma unroll
      for (int i = 0; i < 8; ++i) {
        int cch = i * 32 + lane, row = cch >> 2, off = (cch & 3) * 16;
        async_copy16(vloff + row * 64 + off,
                     vbase + ((size_t)row * cS + kb) * 2 + off);
      }
    }
    wait_async();
    __syncthreads();

    // ---- S^T tiles: 32 keys x 16 queries (VGPR=key, lane=query) ----
    v8f sc[2];
#pragma unroll
    for (int t = 0; t < 2; ++t) {
      const _Float16* kr = sK + (16 * t + ln) * 64;
      v16h A0 = cat8(ldh8(kr + hl * 8),      ldh8(kr + 16 + hl * 8));
      v16h A1 = cat8(ldh8(kr + 32 + hl * 8), ldh8(kr + 48 + hl * 8));
      v8f cc;
#pragma unroll
      for (int r = 0; r < 8; ++r) cc[r] = 0.f;
      cc = wmma_f16(A0, Bq[0], cc);
      cc = wmma_f16(A1, Bq[1], cc);
      sc[t] = cc;
    }

    // ---- masks: this lane's 16 keys = {kb+8*hl+r} u {kb+16+8*hl+r} ----
    unsigned long long m0b = *(const unsigned long long*)(kpr + kb + 8 * hl) |
                             *(const unsigned long long*)(amr + kb + 8 * hl);
    unsigned long long m1b = *(const unsigned long long*)(kpr + kb + 16 + 8 * hl) |
                             *(const unsigned long long*)(amr + kb + 16 + 8 * hl);

    float sv[16];
#pragma unroll
    for (int r = 0; r < 8; ++r) {
      float s0 = sc[0][r] * 0.125f;            // 1/sqrt(DK)
      float s1 = sc[1][r] * 0.125f;
      if ((m0b >> (8 * r)) & 0xffull) s0 = -1e9f;
      if ((m1b >> (8 * r)) & 0xffull) s1 = -1e9f;
      sv[r] = s0; sv[8 + r] = s1;
    }

    // ---- online softmax (per query column) ----
    float mloc = sv[0];
#pragma unroll
    for (int r = 1; r < 16; ++r) mloc = fmaxf(mloc, sv[r]);
    mloc = fmaxf(mloc, __shfl_xor(mloc, 16, 32));
    const float mnew  = fmaxf(mi, mloc);
    const float alpha = __expf(mi - mnew);

    float psum = 0.f;
    v16h Pa;                                   // P A-frag: layout-exact
#pragma unroll
    for (int r = 0; r < 16; ++r) {
      float p = __expf(sv[r] - mnew);
      psum += p;
      Pa[r] = (_Float16)p;
    }
    psum += __shfl_xor(psum, 16, 32);
    li = li * alpha + psum;
    mi = mnew;

    float af[8];
#pragma unroll
    for (int r = 0; r < 8; ++r) af[r] = __shfl(alpha, r + 8 * hl, 32);
#pragma unroll
    for (int f = 0; f < 4; ++f)
#pragma unroll
      for (int r = 0; r < 8; ++r) o[f][r] *= af[r];

    // ---- P·V: B-frags from LDS V (rows=keys, cols=dk) ----
#pragma unroll
    for (int f = 0; f < 4; ++f) {
      const _Float16* vr = sV + (16 * f + ln) * 32;
      v16h Bv = cat8(ldh8(vr + hl * 8), ldh8(vr + 16 + hl * 8));
      o[f] = wmma_f16(Pa, Bv, o[f]);
    }
    __syncthreads();                           // before next overwrite
  }

  float lf[8];
#pragma unroll
  for (int r = 0; r < 8; ++r) lf[r] = 1.f / __shfl(li, r + 8 * hl, 32);
#pragma unroll
  for (int f = 0; f < 4; ++f) {
#pragma unroll
    for (int r = 0; r < 8; ++r) {
      const int row = qb + r + 8 * hl;
      y[((size_t)b * cS + row) * cD + h * cDK + 16 * f + ln] =
          (_Float16)(o[f][r] * lf[r]);
    }
  }
}

// ---------------------------------------------------------------------------
// Output projection: out[m,n] = sum_d y[m,d]*Wout[n,d] + bout[n]  (f32 out)
// Same 32x64 per-wave register tiling; A and B both async-DMA'd into LDS.
// ---------------------------------------------------------------------------
__global__ __launch_bounds__(256) void outproj_kernel(const _Float16* __restrict__ y,
                                                      const _Float16* __restrict__ Wh,
                                                      const float* __restrict__ bout,
                                                      float* __restrict__ outp) {
  __shared__ _Float16 sA[32 * 32];            // 2 KB
  __shared__ _Float16 sB[8 * 4 * 16 * 32];    // 32 KB

  const int tid  = threadIdx.x;
  const int wv   = tid >> 5;
  const int lane = tid & 31;
  const int hl   = lane >> 4;
  const int ln   = lane & 15;
  const int mt   = blockIdx.x >> 1;
  const int ngrp = blockIdx.x & 1;
  const int m0   = mt * 32;
  const int n0   = ngrp * 512 + wv * 64;

  const unsigned int aoff = lds_off_of(sA);
  const unsigned int boff = lds_off_of(sB) + wv * 4096;
  _Float16* sBw = sB + wv * 2048;

  const char* yb = (const char*)y;
  const char* wb = (const char*)Wh;

  v8f c[2][4];
#pragma unroll
  for (int mi = 0; mi < 2; ++mi)
#pragma unroll
    for (int j = 0; j < 4; ++j)
#pragma unroll
      for (int i = 0; i < 8; ++i) c[mi][j][i] = 0.f;

  for (int k0 = 0; k0 < cD; k0 += 32) {
    __syncthreads();
    if (wv == 0) {                       // stage A: 32 rows x 64B (async)
#pragma unroll
      for (int i = 0; i < 4; ++i) {
        int cch = i * 32 + lane, row = cch >> 2, off = (cch & 3) * 16;
        async_copy16(aoff + row * 64 + off,
                     yb + ((size_t)(m0 + row) * cD + k0) * 2 + off);
      }
    }
#pragma unroll
    for (int j = 0; j < 4; ++j)          // stage B: per-wave slice (async)
#pragma unroll
      for (int i = 0; i < 2; ++i) {
        int cch = i * 32 + lane, row = cch >> 2, off = (cch & 3) * 16;
        async_copy16(boff + j * 1024 + row * 64 + off,
                     wb + ((size_t)(n0 + j * 16 + row) * cD + k0) * 2 + off);
      }
    wait_async();
    __syncthreads();

    v16h Af[2], Bf[4];
#pragma unroll
    for (int mi = 0; mi < 2; ++mi) {
      const _Float16* ar = sA + (mi * 16 + ln) * 32;
      Af[mi] = cat8(ldh8(ar + hl * 8), ldh8(ar + 16 + hl * 8));
    }
#pragma unroll
    for (int j = 0; j < 4; ++j) {
      const _Float16* br = sBw + j * 512 + ln * 32;
      Bf[j] = cat8(ldh8(br + hl * 8), ldh8(br + 16 + hl * 8));
    }
#pragma unroll
    for (int mi = 0; mi < 2; ++mi)
#pragma unroll
      for (int j = 0; j < 4; ++j)
        c[mi][j] = wmma_f16(Af[mi], Bf[j], c[mi][j]);
  }

#pragma unroll
  for (int j = 0; j < 4; ++j) {
    const float bv = bout[n0 + j * 16 + ln];
#pragma unroll
    for (int mi = 0; mi < 2; ++mi) {
#pragma unroll
      for (int r = 0; r < 8; ++r) {
        const int m = m0 + mi * 16 + r + 8 * hl;
        outp[(size_t)m * cD + n0 + j * 16 + ln] = c[mi][j][r] + bv;
      }
    }
  }
}

// ---------------------------------------------------------------------------
extern "C" void kernel_launch(void* const* d_in, const int* in_sizes, int n_in,
                              void* d_out, int out_size, void* d_ws, size_t ws_size,
                              hipStream_t stream) {
  const float* q  = (const float*)d_in[0];
  const float* k  = (const float*)d_in[1];
  const float* v  = (const float*)d_in[2];
  const unsigned char* kpm = (const unsigned char*)d_in[3];
  const unsigned char* am  = (const unsigned char*)d_in[4];
  const float* Wq = (const float*)d_in[5];
  const float* bq = (const float*)d_in[6];
  const float* Wk = (const float*)d_in[7];
  const float* bk = (const float*)d_in[8];
  const float* Wv = (const float*)d_in[9];
  const float* bv = (const float*)d_in[10];
  const float* Wout = (const float*)d_in[11];
  const float* bout = (const float*)d_in[12];

  const size_t NN = (size_t)cB * cH * cS * cDK;  // 8388608 (16 MB as f16)
  const size_t NW = (size_t)cD * cD;             // 1048576 (2 MB as f16)
  _Float16* ws  = (_Float16*)d_ws;
  _Float16* qh  = ws;                  // [B,H,S,DK]
  _Float16* kh  = ws + NN;             // [B,H,S,DK]
  _Float16* vt  = ws + 2 * NN;         // [B,H,DK,S]
  _Float16* yy  = ws + 3 * NN;         // [B,S,D]
  _Float16* whq = ws + 4 * NN;         // f16 weights
  _Float16* whk = whq + NW;
  _Float16* whv = whk + NW;
  _Float16* who = whv + NW;

  dim3 blk(256);
  const int n8 = (int)(NW / 8);        // 131072
  cvt_kernel<<<512, blk, 0, stream>>>(Wq, whq, n8);
  cvt_kernel<<<512, blk, 0, stream>>>(Wk, whk, n8);
  cvt_kernel<<<512, blk, 0, stream>>>(Wv, whv, n8);
  cvt_kernel<<<512, blk, 0, stream>>>(Wout, who, n8);

  // 256 M-blocks x 2 N-groups = 512 blocks (each wave: 32x64 output tile)
  proj16_kernel<<<512, blk, 0, stream>>>(q, whq, bq, qh, 0);
  proj16_kernel<<<512, blk, 0, stream>>>(k, whk, bk, kh, 0);
  proj16_kernel<<<512, blk, 0, stream>>>(v, whv, bv, vt, 1);   // V transposed
  flash_attn_kernel<<<1024, blk, 0, stream>>>(qh, kh, vt, kpm, am, yy);
  outproj_kernel<<<512, blk, 0, stream>>>(yy, who, bout, (float*)d_out);
}
